// CausalSelfAttention_20469814132940
// MI455X (gfx1250) — matrix operand order repaired and checked by reading, not compiled, not run on hardware
//
#include <hip/hip_runtime.h>

typedef __attribute__((ext_vector_type(16))) __bf16 v16bf;
typedef __attribute__((ext_vector_type(8)))  float  v8f;

// Problem constants (B=2, T=2048, C=1024, H=16, hd=64)
#define BATCH 2
#define SEQ   2048
#define CDIM  1024
#define NHEAD 16
#define HDIM  64

// ---------------------------------------------------------------------------
// CDNA5 WMMA 16x16x32 bf16 fragment layouts (wave32):
// A 16x32: lane L<16: row m=L, k={0..7,16..23}; lane L>=16: k={8..15,24..31}
// B 32x16: lane L holds col n=L%16, k in the same order as A
// C/D f32 16x16: lane L holds col n=L%16, VGPR r holds row r+8*(L/16)
// ---------------------------------------------------------------------------

// gfx1250 async global->LDS copy (ASYNCcnt-tracked), GV addressing mode.
__device__ __forceinline__ void async_copy_b128(unsigned lds_byte_off,
                                                const void* gptr) {
  asm volatile("global_load_async_to_lds_b128 %0, %1, off"
               :: "v"(lds_byte_off),
                  "v"((unsigned long long)(uintptr_t)gptr)
               : "memory");
}

// ========================== Kernel 1: QKV GEMM =============================
// X[4096x1024] (f32) @ Wqkv[1024x3072] (f32) + bqkv ->
//   Q,V: bf16 [B*T x 1024]; K: bf16 transposed per head KT[b][h][d][t]
__global__ __launch_bounds__(128) void qkv_gemm_kernel(
    const float* __restrict__ A, const float* __restrict__ Bm,
    const float* __restrict__ bias,
    __bf16* __restrict__ Qo, __bf16* __restrict__ KTo, __bf16* __restrict__ Vo) {
  const int Kd = 1024, N = 3072;
  __shared__ __bf16 As[64][32];
  __shared__ __bf16 Bs[32][64];
  const int n0 = blockIdx.x * 64;
  const int m0 = blockIdx.y * 64;
  const int tid = threadIdx.x;
  const int lane = tid & 31, w = tid >> 5;

  v8f acc[4] = {};

  for (int k0 = 0; k0 < Kd; k0 += 32) {
    __syncthreads();
    if (k0 + 32 < Kd) {  // prefetch next tiles into L2 (global_prefetch_b8)
      __builtin_prefetch(&A[(size_t)(m0 + (tid & 63)) * Kd + k0 + 32], 0, 1);
      __builtin_prefetch(&Bm[(size_t)(k0 + 32 + (tid & 31)) * N + n0 + ((tid >> 5) << 4)], 0, 1);
    }
#pragma unroll
    for (int i = 0; i < 16; ++i) {       // A tile 64x32, f32 -> bf16
      int idx = tid * 16 + i;
      int r = idx >> 5, c = idx & 31;
      As[r][c] = (__bf16)A[(size_t)(m0 + r) * Kd + k0 + c];
    }
#pragma unroll
    for (int i = 0; i < 16; ++i) {       // B tile 32x64, f32 -> bf16
      int idx = tid * 16 + i;
      int r = idx >> 6, c = idx & 63;
      Bs[r][c] = (__bf16)Bm[(size_t)(k0 + r) * N + n0 + c];
    }
    __syncthreads();

    v16bf af;
    {
      int m = lane & 15, kb = (lane < 16) ? 0 : 8;
#pragma unroll
      for (int j = 0; j < 8; ++j) {
        af[j]     = As[16 * w + m][kb + j];
        af[8 + j] = As[16 * w + m][kb + 16 + j];
      }
    }
#pragma unroll
    for (int ns = 0; ns < 4; ++ns) {
      v16bf bfb;
#pragma unroll
      for (int j = 0; j < 16; ++j) bfb[j] = Bs[(lane >> 4) * 8 + (j & 7) + ((j >> 3) << 4)][ns * 16 + (lane & 15)];
      acc[ns] = __builtin_amdgcn_wmma_f32_16x16x32_bf16(
          false, af, false, bfb, (short)0, acc[ns], false, false);
    }
  }

  // epilogue: bias + route; K slice stored transposed per head
  const int half = lane >> 4, nn = lane & 15;
#pragma unroll
  for (int ns = 0; ns < 4; ++ns) {
    int col = n0 + ns * 16 + nn;
    float bv = bias[col];
    int slice = (n0 + ns * 16) >> 10;    // wave-uniform per ns
    int c2 = col & 1023;
#pragma unroll
    for (int r = 0; r < 8; ++r) {
      int m = m0 + 16 * w + r + 8 * half;
      float val = acc[ns][r] + bv;
      if (slice == 0) {
        Qo[(size_t)m * 1024 + c2] = (__bf16)val;
      } else if (slice == 2) {
        Vo[(size_t)m * 1024 + c2] = (__bf16)val;
      } else {
        int hh = c2 >> 6, d = c2 & 63;
        int bb = m >> 11, t = m & 2047;
        KTo[(((size_t)bb * NHEAD + hh) * HDIM + d) * SEQ + t] = (__bf16)val;
      }
    }
  }
}

// ====================== Kernel 2: Causal flash attention ===================
// Per (b,h): Q,V row-major [2048 x 64]; KT row-major [64 x 2048] (all bf16).
// Block = 128 threads (4 waves); dynamic LDS so async-copy dests are raw offsets:
//   [0,4096)      KsT[64][32]  (d-major K tile)
//   [4096,8192)   Vs[32][64]
//   [8192,12288)  Ps[4][16][32] per-wave P transpose scratch
#define LDS_KT 0u
#define LDS_V  4096u
#define LDS_P  8192u

__global__ __launch_bounds__(128) void attn_kernel(
    const __bf16* __restrict__ Q, const __bf16* __restrict__ KT,
    const __bf16* __restrict__ V, __bf16* __restrict__ Y) {
  extern __shared__ char smem[];
  const int qb = blockIdx.x;       // 0..31  (query block of 64)
  const int h  = blockIdx.y;       // 0..15
  const int b  = blockIdx.z;       // 0..1
  const int tid = threadIdx.x;
  const int lane = tid & 31, w = tid >> 5;

  const size_t hdbase = ((size_t)b * NHEAD + h) * (size_t)SEQ * HDIM;
  const __bf16* Qh  = Q + hdbase;         // [T][64]
  const __bf16* KTh = KT + hdbase;        // [64][T]
  const __bf16* Vh  = V + hdbase;         // [T][64]
  __bf16* KsT = (__bf16*)(smem + LDS_KT); // [64][32]
  __bf16* Vs  = (__bf16*)(smem + LDS_V);  // [32][64]
  __bf16* Ps  = (__bf16*)(smem + LDS_P) + w * 512;  // [16][32]
  const int qt0 = qb * 64 + w * 16;

  // Q fragments: 16x64 as two 16x32 A-operands held in VGPRs
  v16bf qf[2];
  {
    int m = lane & 15, kb = (lane < 16) ? 0 : 8;
    const __bf16* qrow = Qh + (size_t)(qt0 + m) * HDIM;
#pragma unroll
    for (int c = 0; c < 2; ++c)
#pragma unroll
      for (int j = 0; j < 8; ++j) {
        qf[c][j]     = qrow[c * 32 + kb + j];
        qf[c][8 + j] = qrow[c * 32 + kb + 16 + j];
      }
  }

  v8f acc[4] = {};
  float mstat[8], lstat[8];
#pragma unroll
  for (int r = 0; r < 8; ++r) { mstat[r] = -1e30f; lstat[r] = 0.f; }

  const int half = lane >> 4, nn = lane & 15;
  const int nkt = qb * 2 + 2;

  for (int kt = 0; kt < nkt; ++kt) {
    __syncthreads();   // previous tile fully consumed before overwrite
    // Async global->LDS tile copies: 16B chunks, 2 per thread per tile.
#pragma unroll
    for (int cc = 0; cc < 2; ++cc) {
      int c = tid + cc * 128;            // KT tile: 64 rows x 32 cols bf16
      int row = c >> 2, col8 = (c & 3) * 8;
      async_copy_b128(LDS_KT + (unsigned)(row * 64 + col8 * 2),
                      KTh + (size_t)row * SEQ + kt * 32 + col8);
    }
#pragma unroll
    for (int cc = 0; cc < 2; ++cc) {
      int c = tid + cc * 128;            // V tile: 32 rows x 64 cols bf16
      int row = c >> 3, col8 = (c & 7) * 8;
      async_copy_b128(LDS_V + (unsigned)(row * 128 + col8 * 2),
                      Vh + (size_t)(kt * 32 + row) * HDIM + col8);
    }
    asm volatile("s_wait_asynccnt 0" ::: "memory");
    __syncthreads();

    // S = Q K^T : two 16x16 tiles, d accumulated in 2 chunks of 32
    v8f s[2] = {};
#pragma unroll
    for (int t = 0; t < 2; ++t)
#pragma unroll
      for (int c = 0; c < 2; ++c) {
        v16bf kf;                        // key column n=lane%16, d in table order from KsT
        const __bf16* kp = KsT + (c * 32) * 32 + t * 16 + (lane & 15);
#pragma unroll
        for (int j = 0; j < 16; ++j) kf[j] = kp[((lane >> 4) * 8 + (j & 7) + ((j >> 3) << 4)) * 32];
        s[t] = __builtin_amdgcn_wmma_f32_16x16x32_bf16(
            false, qf[c], false, kf, (short)0, s[t], false, false);
      }

    // causal mask + online softmax (xor-shfl stays within 16-lane halves)
#pragma unroll
    for (int r = 0; r < 8; ++r) {
      int qrow = qt0 + r + 8 * half;
#pragma unroll
      for (int t = 0; t < 2; ++t) {
        int key = kt * 32 + t * 16 + nn;
        float sv = s[t][r] * 0.125f;     // 1/sqrt(64)
        s[t][r] = (key <= qrow) ? sv : -1e30f;
      }
      float localmax = fmaxf(s[0][r], s[1][r]);
#pragma unroll
      for (int off = 1; off < 16; off <<= 1)
        localmax = fmaxf(localmax, __shfl_xor(localmax, off, 32));
      float mnew = fmaxf(mstat[r], localmax);
      float psum = 0.f;
#pragma unroll
      for (int t = 0; t < 2; ++t) {
        float p = __expf(s[t][r] - mnew);
        s[t][r] = p;
        psum += p;
      }
#pragma unroll
      for (int off = 1; off < 16; off <<= 1)
        psum += __shfl_xor(psum, off, 32);
      float scale = __expf(mstat[r] - mnew);
      lstat[r] = lstat[r] * scale + psum;
      mstat[r] = mnew;
#pragma unroll
      for (int d = 0; d < 4; ++d) acc[d][r] *= scale;
#pragma unroll
      for (int t = 0; t < 2; ++t)        // stash P for lane transpose
        Ps[(r + 8 * half) * 32 + t * 16 + nn] = (__bf16)s[t][r];
    }

    asm volatile("s_wait_dscnt 0" ::: "memory");  // wave-local LDS RAW fence

    v16bf pf;                            // P as 16x32 A fragment
    {
      int m = lane & 15, kb = (lane < 16) ? 0 : 8;
#pragma unroll
      for (int j = 0; j < 8; ++j) {
        pf[j]     = Ps[m * 32 + kb + j];
        pf[8 + j] = Ps[m * 32 + kb + 16 + j];
      }
    }
#pragma unroll
    for (int d = 0; d < 4; ++d) {        // acc += P @ V
      v16bf vfv;
      const __bf16* vp = Vs + d * 16 + (lane & 15);
#pragma unroll
      for (int j = 0; j < 16; ++j) vfv[j] = vp[((lane >> 4) * 8 + (j & 7) + ((j >> 3) << 4)) * 64];
      acc[d] = __builtin_amdgcn_wmma_f32_16x16x32_bf16(
          false, pf, false, vfv, (short)0, acc[d], false, false);
    }
  }

  // normalize and store in (B,T,C) transposed-head layout for proj GEMM
#pragma unroll
  for (int r = 0; r < 8; ++r) {
    float inv = 1.0f / lstat[r];
    int t = qt0 + r + 8 * half;
#pragma unroll
    for (int d = 0; d < 4; ++d) {
      int dd = d * 16 + nn;
      Y[((size_t)(b * SEQ + t)) * CDIM + h * HDIM + dd] = (__bf16)(acc[d][r] * inv);
    }
  }
}

// ========================== Kernel 3: Proj GEMM ============================
// Ybf[4096x1024] (bf16) @ Wproj[1024x1024] (f32->bf16) + bproj -> out f32
__global__ __launch_bounds__(128) void proj_gemm_kernel(
    const __bf16* __restrict__ A, const float* __restrict__ Bm,
    const float* __restrict__ bias, float* __restrict__ out) {
  const int Kd = 1024, N = 1024;
  __shared__ __bf16 As[64][32];
  __shared__ __bf16 Bs[32][64];
  const int n0 = blockIdx.x * 64;
  const int m0 = blockIdx.y * 64;
  const int tid = threadIdx.x;
  const int lane = tid & 31, w = tid >> 5;

  v8f acc[4] = {};

  for (int k0 = 0; k0 < Kd; k0 += 32) {
    __syncthreads();
    if (k0 + 32 < Kd) {
      __builtin_prefetch(&A[(size_t)(m0 + (tid & 63)) * Kd + k0 + 32], 0, 1);
      __builtin_prefetch(&Bm[(size_t)(k0 + 32 + (tid & 31)) * N + n0 + ((tid >> 5) << 4)], 0, 1);
    }
#pragma unroll
    for (int i = 0; i < 16; ++i) {
      int idx = tid * 16 + i;
      int r = idx >> 5, c = idx & 31;
      As[r][c] = A[(size_t)(m0 + r) * Kd + k0 + c];
    }
#pragma unroll
    for (int i = 0; i < 16; ++i) {
      int idx = tid * 16 + i;
      int r = idx >> 6, c = idx & 63;
      Bs[r][c] = (__bf16)Bm[(size_t)(k0 + r) * N + n0 + c];
    }
    __syncthreads();

    v16bf af;
    {
      int m = lane & 15, kb = (lane < 16) ? 0 : 8;
#pragma unroll
      for (int j = 0; j < 8; ++j) {
        af[j]     = As[16 * w + m][kb + j];
        af[8 + j] = As[16 * w + m][kb + 16 + j];
      }
    }
#pragma unroll
    for (int ns = 0; ns < 4; ++ns) {
      v16bf bfb;
#pragma unroll
      for (int j = 0; j < 16; ++j) bfb[j] = Bs[(lane >> 4) * 8 + (j & 7) + ((j >> 3) << 4)][ns * 16 + (lane & 15)];
      acc[ns] = __builtin_amdgcn_wmma_f32_16x16x32_bf16(
          false, af, false, bfb, (short)0, acc[ns], false, false);
    }
  }

  const int half = lane >> 4, nn = lane & 15;
#pragma unroll
  for (int ns = 0; ns < 4; ++ns) {
    int col = n0 + ns * 16 + nn;
    float bv = bias[col];
#pragma unroll
    for (int r = 0; r < 8; ++r) {
      int m = m0 + 16 * w + r + 8 * half;
      out[(size_t)m * N + col] = acc[ns][r] + bv;
    }
  }
}

// ================================ Launch ===================================
extern "C" void kernel_launch(void* const* d_in, const int* in_sizes, int n_in,
                              void* d_out, int out_size, void* d_ws, size_t ws_size,
                              hipStream_t stream) {
  const float* x     = (const float*)d_in[0];
  const float* Wqkv  = (const float*)d_in[1];
  const float* bqkv  = (const float*)d_in[2];
  const float* Wproj = (const float*)d_in[3];
  const float* bproj = (const float*)d_in[4];
  float* out = (float*)d_out;

  // workspace: Qbf | KTbf | Vbf | Ybf, each B*T*C bf16 = 8 MiB (32 MiB total)
  const size_t SLICE = (size_t)BATCH * SEQ * CDIM * sizeof(__bf16);
  char* ws = (char*)d_ws;
  __bf16* Qbf  = (__bf16*)(ws);
  __bf16* KTbf = (__bf16*)(ws + SLICE);
  __bf16* Vbf  = (__bf16*)(ws + 2 * SLICE);
  __bf16* Ybf  = (__bf16*)(ws + 3 * SLICE);

  // 1) x @ Wqkv + bqkv -> Q / KT / V (bf16)
  qkv_gemm_kernel<<<dim3(3072 / 64, (BATCH * SEQ) / 64), 128, 0, stream>>>(
      x, Wqkv, bqkv, Qbf, KTbf, Vbf);

  // 2) causal attention per (b, h, 64-row query block); 12KB dynamic LDS
  attn_kernel<<<dim3(SEQ / 64, NHEAD, BATCH), 128, 12288, stream>>>(
      Qbf, KTbf, Vbf, Ybf);

  // 3) y @ Wproj + bproj -> out (f32)
  proj_gemm_kernel<<<dim3(CDIM / 64, (BATCH * SEQ) / 64), 128, 0, stream>>>(
      Ybf, Wproj, bproj, out);
}
